// NeuralCausalModel_13348758356448
// MI455X (gfx1250) — matrix operand order repaired and checked by reading, not compiled, not run on hardware
//
#include <hip/hip_runtime.h>

typedef __attribute__((ext_vector_type(16))) _Float16 v16h;
typedef __attribute__((ext_vector_type(8)))  _Float16 v8h;
typedef __attribute__((ext_vector_type(8)))  float    v8f;

#define HID 64
#define TILE_M 16

static __device__ __forceinline__ v16h splat16(_Float16 x) {
    v16h r;
#pragma unroll
    for (int i = 0; i < 16; ++i) r[i] = x;
    return r;
}

// Single-instruction ReLU: v_med3_f32(x, 0, FLT_MAX) == clamp-below-0 for all
// finite inputs; avoids the canonicalize+max pair the backend emits for maxnum
// on (opaque) WMMA results.
static __device__ __forceinline__ float relu1(float x) {
    return __builtin_amdgcn_fmed3f(x, 0.0f, 3.402823466e38f);
}

// One variable of the causal chain, 16 samples per wave-tile.
// Operand-swapped WMMA:  D = W2^T (A, weight-stationary) x h1 (B, per-tile).
// Layer 1 in packed f16 straight into the B-fragment registers; layer 3 as a
// v8f packed accumulate (hidden index lives on r/half/t), one shuffle, one
// coalesced 16-lane store.
__global__ __launch_bounds__(256, 4) void ncm_var_kernel(
    const float* __restrict__ parent,      // previous variable's samples [N]
    const float* __restrict__ noise_v,     // this variable's noise [N]
    const float* __restrict__ noise_stds,  // [V]
    int v,
    const float* __restrict__ w1a,   // [64] coeffs for input slot 0
    const float* __restrict__ w1b,   // [64] coeffs for input slot 1
    const float* __restrict__ b1,    // [64]
    const float* __restrict__ w2,    // [64][64] row-major [k][n]
    const float* __restrict__ b2,    // [64]
    const float* __restrict__ w3,    // [64]
    const float* __restrict__ b3,    // [1]
    float* __restrict__ out,         // [N]
    int nsamples, int use_parent)
{
    __shared__ __align__(32) _Float16 sW2T[HID * HID];  // transposed: [n][k], f16
    __shared__ __align__(32) _Float16 sW0h[HID], sW1h[HID], sB1h[HID];
    __shared__ __align__(32) float sB2[HID], sW3[HID];

    const int tid = threadIdx.x;

    // Stage weights; W2 transposed so A-fragment rows (n-major) are contiguous.
    for (int i = tid; i < HID * HID; i += blockDim.x) {
        int k = i >> 6, n = i & 63;
        sW2T[n * HID + k] = (_Float16)w2[i];
    }
    if (tid < HID) {
        sW0h[tid] = (_Float16)w1a[tid];
        sW1h[tid] = use_parent ? (_Float16)w1b[tid] : (_Float16)0.0f;  // root: zero slot 1
        sB1h[tid] = (_Float16)b1[tid];
        sB2[tid]  = b2[tid];
        sW3[tid]  = w3[tid];
    }
    __syncthreads();

    const int lane = tid & 31;
    const int half = lane >> 4;   // which 16-lane half of the wave
    const int lm   = lane & 15;

    // Hoist 8 W2^T A-fragments (4 M(=n)-tiles x 2 K-halves) into registers.
    // A layout used here (contiguous k): lane half h, row M=lm, K runs [32f+16h, +16) (matches B).
    v16h afrag[4][2];
#pragma unroll
    for (int t = 0; t < 4; ++t)
#pragma unroll
        for (int f = 0; f < 2; ++f) {
            const _Float16* row = &sW2T[(16 * t + lm) * HID];
            v8h lo = *(const v8h*)(row + 32 * f + 16 * half);
            v8h hi = *(const v8h*)(row + 32 * f + 16 * half + 8);
            afrag[t][f] = __builtin_shufflevector(lo, hi,
                0, 1, 2, 3, 4, 5, 6, 7, 8, 9, 10, 11, 12, 13, 14, 15);
        }

    // Hoist layer-1 weight/bias vectors: each lane's k-range is contiguous
    // (B-fragment layout: K = 32f + 16*half + j), so these are single 32B LDS loads.
    v16h w0v[2], w1v[2], b1v[2];
#pragma unroll
    for (int f = 0; f < 2; ++f) {
        const int ks = 32 * f + 16 * half;
        w0v[f] = *(const v16h*)&sW0h[ks];
        w1v[f] = *(const v16h*)&sW1h[ks];
        b1v[f] = *(const v16h*)&sB1h[ks];
    }

    const v16h zeroh = splat16((_Float16)0.0f);

    const float astd = fabsf(noise_stds[v]);
    const float b3v  = b3[0];

    const int ntiles     = nsamples >> 4;                     // 16 samples per wave-tile
    const int wave_id    = (blockIdx.x * blockDim.x + tid) >> 5;
    const int wave_count = (gridDim.x * blockDim.x) >> 5;

    for (int tile = wave_id; tile < ntiles; tile += wave_count) {
        const int s = tile * TILE_M + lm;        // lanes l and l+16 share sample lm

        // prefetch next iteration's operand streams
        const int sn = s + wave_count * TILE_M;
        if (sn < nsamples) {
            __builtin_prefetch(&noise_v[sn], 0, 0);
            if (use_parent) __builtin_prefetch(&parent[sn], 0, 0);
        }

        const float e  = noise_v[s] * astd;
        const float x0 = use_parent ? parent[s] : e;
        const float x1 = e;                       // multiplied by zeroed w1v for the root

        // Layer 1 + ReLU in packed f16, directly in WMMA B-fragment layout.
        const v16h x0v = splat16((_Float16)x0);
        const v16h x1v = splat16((_Float16)x1);
        v16h bfr[2];
#pragma unroll
        for (int f = 0; f < 2; ++f) {
            v16h h = x0v * w0v[f] + (x1v * w1v[f] + b1v[f]);  // v_pk_fma_f16
            bfr[f] = __builtin_elementwise_max(h, zeroh);      // v_pk_max_num_f16
        }

        // Layer 2 (WMMA, C pre-loaded with b2 along the register axis) + ReLU
        // (single v_med3 each), layer 3 as packed v8f accumulate.
        v8f accv = {0.f, 0.f, 0.f, 0.f, 0.f, 0.f, 0.f, 0.f};
#pragma unroll
        for (int t = 0; t < 4; ++t) {
            v8f c = *(const v8f*)&sB2[16 * t + 8 * half];
            c = __builtin_amdgcn_wmma_f32_16x16x32_f16(false, afrag[t][0], false, bfr[0],
                                                       (short)0, c, false, false);
            c = __builtin_amdgcn_wmma_f32_16x16x32_f16(false, afrag[t][1], false, bfr[1],
                                                       (short)0, c, false, false);
            const v8f w3v = *(const v8f*)&sW3[16 * t + 8 * half];
            v8f cr;
#pragma unroll
            for (int r = 0; r < 8; ++r)
                cr[r] = relu1(c[r]);                           // v_med3_f32
            accv = cr * w3v + accv;                            // v_pk_fma_f32 x4
        }

        // Horizontal sum (balanced tree), then combine the two K-halves.
        float acc = ((accv[0] + accv[4]) + (accv[1] + accv[5]))
                  + ((accv[2] + accv[6]) + (accv[3] + accv[7]));
        acc += __shfl_xor(acc, 16, 32);

        if (half == 0)
            out[tile * TILE_M + lm] = acc + b3v;   // coalesced 16-lane store
    }
}

extern "C" void kernel_launch(void* const* d_in, const int* in_sizes, int n_in,
                              void* d_out, int out_size, void* d_ws, size_t ws_size,
                              hipStream_t stream) {
    const int V = 16;
    const float* noise = (const float*)d_in[0];   // [V,N,1]
    const float* stds  = (const float*)d_in[1];   // [V]
    const float* w0_1  = (const float*)d_in[2];   // [1,64]
    const float* b0_1  = (const float*)d_in[3];   // [64]
    const float* w0_2  = (const float*)d_in[4];   // [64,64]
    const float* b0_2  = (const float*)d_in[5];   // [64]
    const float* w0_3  = (const float*)d_in[6];   // [64,1]
    const float* b0_3  = (const float*)d_in[7];   // [1]
    const float* W1    = (const float*)d_in[8];   // [15,2,64]
    const float* B1    = (const float*)d_in[9];   // [15,64]
    const float* W2    = (const float*)d_in[10];  // [15,64,64]
    const float* B2    = (const float*)d_in[11];  // [15,64]
    const float* W3    = (const float*)d_in[12];  // [15,64,1]
    const float* B3    = (const float*)d_in[13];  // [15,1]
    float* out = (float*)d_out;                   // [V,N,1] fp32

    const int N = in_sizes[0] / V;

    dim3 grid(1024), block(256);  // 8192 waves, grid-stride over N/16 = 32768 tiles

    // Root variable: input is eps only (slot-1 weights zeroed inside the kernel).
    ncm_var_kernel<<<grid, block, 0, stream>>>(
        noise, noise, stds, 0,
        w0_1, w0_1, b0_1, w0_2, b0_2, w0_3, b0_3,
        out, N, 0);

    // Chain: variable v consumes variable v-1's samples; stream order serializes.
    for (int v = 1; v < V; ++v) {
        const size_t u = (size_t)(v - 1);
        ncm_var_kernel<<<grid, block, 0, stream>>>(
            out + u * N, noise + (size_t)v * N, stds, v,
            W1 + u * 2 * 64, W1 + u * 2 * 64 + 64,
            B1 + u * 64,
            W2 + u * 64 * 64, B2 + u * 64,
            W3 + u * 64, B3 + u,
            out + (size_t)v * N, N, 1);
    }
}
